// TransformerEncoder_31112743092512
// MI455X (gfx1250) — matrix operand order repaired and checked by reading, not compiled, not run on hardware
//
#include <hip/hip_runtime.h>
#include <hip/hip_bf16.h>

typedef _Float16 h16;
typedef __attribute__((ext_vector_type(16))) _Float16 v16h;
typedef __attribute__((ext_vector_type(8)))  _Float16 v8h;
typedef __attribute__((ext_vector_type(8)))  float    v8f;

#define BDIM 1024
#define FFDIM 4096
#define SEQ 2048
#define NB 4
#define NH 16
#define MTOK 8192  // NB*SEQ

// ---- CDNA5 async global->LDS copy (ASYNCcnt-tracked DMA, no VGPR round trip) ----
__device__ inline void async_copy_b128(const h16* g, h16* l) {
    unsigned lds_off = (unsigned)(uintptr_t)l;           // low 32 bits = LDS offset
    unsigned long long ga = (unsigned long long)(uintptr_t)g;
    asm volatile("global_load_async_to_lds_b128 %0, %1, off"
                 :: "v"(lds_off), "v"(ga) : "memory");
}
__device__ inline void async_wait0() { asm volatile("s_wait_asynccnt 0" ::: "memory"); }
__device__ inline void async_wait4() { asm volatile("s_wait_asynccnt 4" ::: "memory"); }

// ---- DPP16 16-lane xor-butterfly reductions (VALU-only; keeps LDS pipe free for WMMA feeds) ----
template <int CTRL>
__device__ inline float dpp_movf(float x) {
    int r = __builtin_amdgcn_update_dpp(0, __float_as_int(x), CTRL, 0xf, 0xf, true);
    return __int_as_float(r);
}
__device__ inline float red16_max(float v) {
    v = fmaxf(v, dpp_movf<0xB1>(v));    // quad_perm [1,0,3,2]  : xor 1
    v = fmaxf(v, dpp_movf<0x4E>(v));    // quad_perm [2,3,0,1]  : xor 2
    v = fmaxf(v, dpp_movf<0x141>(v));   // row_half_mirror      : xor 4 (quads uniform)
    v = fmaxf(v, dpp_movf<0x140>(v));   // row_mirror           : xor 8 (octets uniform)
    return v;
}
__device__ inline float red16_sum(float v) {
    v += dpp_movf<0xB1>(v);
    v += dpp_movf<0x4E>(v);
    v += dpp_movf<0x141>(v);
    v += dpp_movf<0x140>(v);
    return v;
}

// Load a 16x32 f16 A-fragment (contiguous K order, matching B):
// lane<16:  halves 0..15 -> K 0..15
// lane>=16: halves 0..15 -> K 16..31
__device__ inline v16h ld_afrag(const h16* base, int koffA) {
    v8h lo = *(const v8h*)(base + 2 * koffA);
    v8h hi = *(const v8h*)(base + 2 * koffA + 8);
    return __builtin_shufflevector(lo, hi, 0,1,2,3,4,5,6,7,8,9,10,11,12,13,14,15);
}

// ---------------- weight conversion / packing (weights stored TRANSPOSED: [N][K]) ----------------
__global__ void __launch_bounds__(256) pack_qkv_w(const float* __restrict__ Wq,
                                                  const float* __restrict__ Wk,
                                                  const float* __restrict__ Wv,
                                                  h16* __restrict__ dst) {
    int i = blockIdx.x * 256 + threadIdx.x;   // over 3072*1024, dst[n][k]
    int n = i / 1024, k = i % 1024;
    float v = (n < 1024) ? Wq[k * 1024 + n]
            : (n < 2048) ? Wk[k * 1024 + (n - 1024)]
                         : Wv[k * 1024 + (n - 2048)];
    dst[i] = (h16)v;
}

__global__ void __launch_bounds__(256) pack_qkv_b(const float* __restrict__ bq,
                                                  const float* __restrict__ bk,
                                                  const float* __restrict__ bv,
                                                  float* __restrict__ dst) {
    int n = blockIdx.x * 256 + threadIdx.x;   // over 3072
    dst[n] = (n < 1024) ? bq[n] : (n < 2048) ? bk[n - 1024] : bv[n - 2048];
}

// transpose-convert: src f32 [K][N] -> dst f16 [N][K]
__global__ void __launch_bounds__(256) cvt_t_f32_f16(const float* __restrict__ s,
                                                     h16* __restrict__ d, int K, int N) {
    int i = blockIdx.x * 256 + threadIdx.x;   // dst-linear: i = n*K + k
    int n = i / K, k = i % K;
    d[i] = (h16)s[(size_t)k * N + n];
}

// ---------------- layernorm: fp32 in -> f16 out ----------------
__global__ void __launch_bounds__(256) ln_kernel(const float* __restrict__ X,
                                                 const float* __restrict__ g,
                                                 const float* __restrict__ be,
                                                 h16* __restrict__ out) {
    int row = blockIdx.x, tid = threadIdx.x;
    const float* xr = X + (size_t)row * BDIM;
    float v[4], s = 0.f, sq = 0.f;
#pragma unroll
    for (int i = 0; i < 4; ++i) { v[i] = xr[tid + 256 * i]; s += v[i]; sq += v[i] * v[i]; }
#pragma unroll
    for (int off = 16; off; off >>= 1) { s += __shfl_xor(s, off, 32); sq += __shfl_xor(sq, off, 32); }
    __shared__ float rs[8], rq[8], st[2];
    int wave = tid >> 5, lane = tid & 31;
    if (lane == 0) { rs[wave] = s; rq[wave] = sq; }
    __syncthreads();
    if (tid == 0) {
        float ts = 0.f, tq = 0.f;
        for (int i = 0; i < 8; ++i) { ts += rs[i]; tq += rq[i]; }
        float mu = ts * (1.f / BDIM);
        float var = tq * (1.f / BDIM) - mu * mu;
        st[0] = mu; st[1] = rsqrtf(var + 1e-12f);
    }
    __syncthreads();
    float mu = st[0], rstd = st[1];
#pragma unroll
    for (int i = 0; i < 4; ++i) {
        int c = tid + 256 * i;
        out[(size_t)row * BDIM + c] = (h16)((v[i] - mu) * rstd * g[c] + be[c]);
    }
}

// ---------------- WMMA GEMM: C = A(f16,[M][K]) * B(f16 given TRANSPOSED [N][K]) + bias ----------------
// Double-buffered async global->LDS staging; optional exact GELU / fp32 residual; f32 and/or f16 out.
template <bool GELU>
__global__ void __launch_bounds__(256) gemm_wmma(const h16* __restrict__ A,
                                                 const h16* __restrict__ BT,
                                                 const float* __restrict__ bias,
                                                 const float* __restrict__ resid,
                                                 float* __restrict__ outF,
                                                 h16* __restrict__ outH,
                                                 int Ndim, int Kdim) {
    __shared__ __align__(16) h16 As[2][128][32];
    __shared__ __align__(16) h16 Bs[2][128][32];   // Bs[buf][n][k]

    const int tid = threadIdx.x;
    const int wave = tid >> 5, lane = tid & 31;
    const int ln16 = lane & 15, hi = lane >> 4;
    const int koffA = hi ? 8 : 0, koffB = hi ? 16 : 0;
    const int mBase = blockIdx.y * 128, nBase = blockIdx.x * 128;

    // each thread copies 2x16B of A-tile and 2x16B of B-tile per K-step
    const int r0 = tid >> 2, c0 = (tid & 3) * 8;

    auto issue = [&](int buf, int kt) {
        const h16* ga = A + (size_t)(mBase + r0) * Kdim + kt + c0;
        async_copy_b128(ga, &As[buf][r0][c0]);
        async_copy_b128(ga + (size_t)64 * Kdim, &As[buf][r0 + 64][c0]);
        const h16* gb = BT + (size_t)(nBase + r0) * Kdim + kt + c0;
        async_copy_b128(gb, &Bs[buf][r0][c0]);
        async_copy_b128(gb + (size_t)64 * Kdim, &Bs[buf][r0 + 64][c0]);
    };

    const int T = Kdim / 32;
    issue(0, 0);

    v8f acc[8] = {};
    for (int t = 0; t < T; ++t) {
        const int cur = t & 1;
        if (t + 1 < T) { issue(cur ^ 1, (t + 1) * 32); async_wait4(); }
        else           { async_wait0(); }
        __syncthreads();

        v16h a = ld_afrag(&As[cur][wave * 16 + ln16][0], koffA);
#pragma unroll
        for (int nb = 0; nb < 8; ++nb) {
            v16h bf = *(const v16h*)&Bs[cur][nb * 16 + ln16][koffB];
            acc[nb] = __builtin_amdgcn_wmma_f32_16x16x32_f16(
                false, a, false, bf, (short)0, acc[nb], false, false);
        }
        __syncthreads();
    }

#pragma unroll
    for (int nb = 0; nb < 8; ++nb) {
        int col = nBase + nb * 16 + ln16;
        float bs = bias[col];
#pragma unroll
        for (int r = 0; r < 8; ++r) {
            int row = mBase + wave * 16 + r + (hi ? 8 : 0);
            float v = acc[nb][r] + bs;
            if (GELU) v = 0.5f * v * (1.0f + erff(v * 0.70710678118f));
            size_t idx = (size_t)row * Ndim + col;
            if (resid) v += resid[idx];
            if (outF) outF[idx] = v;
            if (outH) outH[idx] = (h16)v;
        }
    }
}

// ---------------- flash attention (per b,h; 128 query rows per WG) ----------------
// qkv layout: qkv[token*3072 + {0|1024|2048} + h*64 + d], token = b*SEQ + l
__global__ void __launch_bounds__(256) attn_kernel(const h16* __restrict__ qkv,
                                                   const int* __restrict__ mask,
                                                   h16* __restrict__ O) {
    __shared__ __align__(16) h16 Ks[32][64];      // 32 keys x 64 dims (async staged)
    __shared__ __align__(16) h16 Vt[64][32];      // transposed: Vt[d][key]
    __shared__ __align__(16) h16 Ps[8][16][32];   // per-wave P scratch

    const int tid = threadIdx.x;
    const int wave = tid >> 5, lane = tid & 31;
    const int ln16 = lane & 15, hi = lane >> 4;
    const int koffA = hi ? 8 : 0, koffB = hi ? 16 : 0;
    const int b = blockIdx.y >> 4, h = blockIdx.y & 15;
    const int qbase = blockIdx.x * 128 + wave * 16;

    // resident Q fragments (16 rows x 64 dims -> two 16x32 A-fragments)
    const h16* qp = qkv + (size_t)(b * SEQ + qbase + ln16) * 3072 + h * 64;
    v16h a0 = ld_afrag(qp, koffA);
    v16h a1 = ld_afrag(qp + 32, koffA);

    v8f oacc[4] = {};
    float mrow[8], lrow[8];
#pragma unroll
    for (int r = 0; r < 8; ++r) { mrow[r] = -1e30f; lrow[r] = 0.f; }

    const int krow = tid >> 3, kc = (tid & 7) * 8;

    for (int kb = 0; kb < SEQ / 32; ++kb) {
        {   // K block: async DMA straight into LDS; V block: load + transpose scatter
            const h16* kp = qkv + (size_t)(b * SEQ + kb * 32 + krow) * 3072 + 1024 + h * 64 + kc;
            async_copy_b128(kp, &Ks[krow][kc]);
            v8h vv = *(const v8h*)(kp + 1024);
#pragma unroll
            for (int i = 0; i < 8; ++i) Vt[kc + i][krow] = vv[i];
        }
        async_wait0();
        __syncthreads();

        // scores: S = Q*K^T, two 16x16 tiles (keys 0..15, 16..31 of this block)
        v8f s0 = {}, s1 = {};
        {
            v16h b00 = *(const v16h*)&Ks[ln16][koffB];
            v16h b01 = *(const v16h*)&Ks[ln16][32 + koffB];
            v16h b10 = *(const v16h*)&Ks[16 + ln16][koffB];
            v16h b11 = *(const v16h*)&Ks[16 + ln16][32 + koffB];
            s0 = __builtin_amdgcn_wmma_f32_16x16x32_f16(false, a0, false, b00, (short)0, s0, false, false);
            s0 = __builtin_amdgcn_wmma_f32_16x16x32_f16(false, a1, false, b01, (short)0, s0, false, false);
            s1 = __builtin_amdgcn_wmma_f32_16x16x32_f16(false, a0, false, b10, (short)0, s1, false, false);
            s1 = __builtin_amdgcn_wmma_f32_16x16x32_f16(false, a1, false, b11, (short)0, s1, false, false);
        }

        const int key0 = kb * 32 + ln16;
        const bool msk0 = mask[b * SEQ + key0] != 0;
        const bool msk1 = mask[b * SEQ + key0 + 16] != 0;

#pragma unroll
        for (int r = 0; r < 8; ++r) {
            float f0 = msk0 ? s0[r] * 0.125f : -1e9f;   // 1/sqrt(64)
            float f1 = msk1 ? s1[r] * 0.125f : -1e9f;
            float mx = red16_max(fmaxf(f0, f1));        // DPP butterfly (VALU only)
            float mn = fmaxf(mrow[r], mx);
            float alpha = __expf(mrow[r] - mn);
            float p0 = __expf(f0 - mn), p1 = __expf(f1 - mn);
            float ps = red16_sum(p0 + p1);              // DPP butterfly (VALU only)
            lrow[r] = lrow[r] * alpha + ps;
            mrow[r] = mn;
#pragma unroll
            for (int db = 0; db < 4; ++db) oacc[db][r] *= alpha;
            int prow = r + (hi ? 8 : 0);
            Ps[wave][prow][ln16] = (h16)p0;
            Ps[wave][prow][16 + ln16] = (h16)p1;
        }
        __builtin_amdgcn_wave_barrier();   // keep DS stores before DS loads (in-order per wave)

        // O += P * V
        v16h pa = ld_afrag(&Ps[wave][ln16][0], koffA);
#pragma unroll
        for (int db = 0; db < 4; ++db) {
            v16h bf = *(const v16h*)&Vt[db * 16 + ln16][koffB];
            oacc[db] = __builtin_amdgcn_wmma_f32_16x16x32_f16(
                false, pa, false, bf, (short)0, oacc[db], false, false);
        }
        __syncthreads();
    }

#pragma unroll
    for (int db = 0; db < 4; ++db) {
        int d = db * 16 + ln16;
#pragma unroll
        for (int r = 0; r < 8; ++r) {
            int row = qbase + r + (hi ? 8 : 0);
            float val = oacc[db][r] / lrow[r];
            O[(size_t)(b * SEQ + row) * BDIM + h * 64 + d] = (h16)val;
        }
    }
}

// ---------------- host orchestration ----------------
extern "C" void kernel_launch(void* const* d_in, const int* in_sizes, int n_in,
                              void* d_out, int out_size, void* d_ws, size_t ws_size,
                              hipStream_t stream) {
    const float* x    = (const float*)d_in[0];
    const int*   mask = (const int*)d_in[1];
    const float* Wq = (const float*)d_in[2];  const float* bq = (const float*)d_in[3];
    const float* Wk = (const float*)d_in[4];  const float* bk = (const float*)d_in[5];
    const float* Wv = (const float*)d_in[6];  const float* bv = (const float*)d_in[7];
    const float* Wo = (const float*)d_in[8];  const float* bo = (const float*)d_in[9];
    const float* W1 = (const float*)d_in[10]; const float* b1 = (const float*)d_in[11];
    const float* W2 = (const float*)d_in[12]; const float* b2 = (const float*)d_in[13];
    const float* g1 = (const float*)d_in[14]; const float* be1 = (const float*)d_in[15];
    const float* g2 = (const float*)d_in[16]; const float* be2 = (const float*)d_in[17];
    float* out = (float*)d_out;

    char* w = (char*)d_ws;
    size_t o_h    = 0;                                   // h/h2 f16   : 16 MiB
    size_t o_wqkv = o_h    + (size_t)MTOK * BDIM * 2;    // WqkvT f16  :  6 MiB
    size_t o_wo   = o_wqkv + (size_t)BDIM * 3 * BDIM * 2;
    size_t o_w1   = o_wo   + (size_t)BDIM * BDIM * 2;
    size_t o_w2   = o_w1   + (size_t)BDIM * FFDIM * 2;
    size_t o_bqkv = o_w2   + (size_t)FFDIM * BDIM * 2;
    size_t o_qkv  = o_bqkv + (size_t)3 * BDIM * 4;       // qkv f16    : 48 MiB
    size_t o_ao   = o_qkv  + (size_t)MTOK * 3 * BDIM * 2;// attnO f16  : 16 MiB
    size_t o_x2   = o_ao   + (size_t)MTOK * BDIM * 2;    // x2 f32     : 32 MiB

    h16*   hf16   = (h16*)(w + o_h);
    h16*   wqkvT  = (h16*)(w + o_wqkv);   // [3072][1024]
    h16*   woT    = (h16*)(w + o_wo);     // [1024][1024]
    h16*   w1T    = (h16*)(w + o_w1);     // [4096][1024]
    h16*   w2T    = (h16*)(w + o_w2);     // [1024][4096]
    float* bqkv   = (float*)(w + o_bqkv);
    h16*   qkv    = (h16*)(w + o_qkv);
    h16*   attnO  = (h16*)(w + o_ao);
    float* x2     = (float*)(w + o_x2);
    h16*   h2f16  = hf16;              // alias (h dead after QKV GEMM)
    h16*   ff1    = qkv;               // alias qkv+attnO region (64 MiB, both dead)

    // weight prep (transposed to [N][K])
    pack_qkv_w<<<(BDIM * 3 * BDIM) / 256, 256, 0, stream>>>(Wq, Wk, Wv, wqkvT);
    pack_qkv_b<<<(3 * BDIM) / 256, 256, 0, stream>>>(bq, bk, bv, bqkv);
    cvt_t_f32_f16<<<(BDIM * BDIM) / 256, 256, 0, stream>>>(Wo, woT, BDIM, BDIM);
    cvt_t_f32_f16<<<(BDIM * FFDIM) / 256, 256, 0, stream>>>(W1, w1T, BDIM, FFDIM);
    cvt_t_f32_f16<<<(FFDIM * BDIM) / 256, 256, 0, stream>>>(W2, w2T, FFDIM, BDIM);

    // h = LN1(x)
    ln_kernel<<<MTOK, 256, 0, stream>>>(x, g1, be1, hf16);

    // qkv = h @ [Wq|Wk|Wv] + [bq|bk|bv]
    gemm_wmma<false><<<dim3(3 * BDIM / 128, MTOK / 128), 256, 0, stream>>>(
        hf16, wqkvT, bqkv, nullptr, nullptr, qkv, 3 * BDIM, BDIM);

    // flash attention
    attn_kernel<<<dim3(SEQ / 128, NB * NH), 256, 0, stream>>>(qkv, mask, attnO);

    // x2 = x + attnO @ Wo + bo
    gemm_wmma<false><<<dim3(BDIM / 128, MTOK / 128), 256, 0, stream>>>(
        attnO, woT, bo, x, x2, nullptr, BDIM, BDIM);

    // h2 = LN2(x2)
    ln_kernel<<<MTOK, 256, 0, stream>>>(x2, g2, be2, h2f16);

    // ff1 = gelu(h2 @ W1 + b1)
    gemm_wmma<true><<<dim3(FFDIM / 128, MTOK / 128), 256, 0, stream>>>(
        h2f16, w1T, b1, nullptr, nullptr, ff1, FFDIM, BDIM);

    // out = x2 + ff1 @ W2 + b2
    gemm_wmma<false><<<dim3(BDIM / 128, MTOK / 128), 256, 0, stream>>>(
        ff1, w2T, b2, x2, out, nullptr, BDIM, FFDIM);
}
